// Head_54357106098770
// MI455X (gfx1250) — compile-verified, matrix operand order checked by reading, not run on hardware
//
#include <hip/hip_runtime.h>
#include <hip/hip_bf16.h>
#include <stdint.h>

typedef unsigned short u16;
typedef unsigned int   u32;

typedef __attribute__((ext_vector_type(16))) __bf16 v16bf;
typedef __attribute__((ext_vector_type(8)))  float  v8f;

struct alignas(16) U4 { u32 x, y, z, w; };
struct alignas(16) AB { U4 lo, hi; };          // 32 bytes -> v16bf
struct alignas(8)  U2 { u32 a, b; };
struct H16 { u16 h[16]; };                     // 32 bytes -> v16bf

// fp32 -> bf16 (round to nearest even, ignoring NaN corner)
__device__ __forceinline__ u16 f2bf(float f) {
    u32 u = __builtin_bit_cast(u32, f);
    u32 r = u + 0x7FFFu + ((u >> 16) & 1u);
    return (u16)(r >> 16);
}
__device__ __forceinline__ u32 f2bf2(float lo, float hi) {
    return (u32)f2bf(lo) | ((u32)f2bf(hi) << 16);
}

// Build a bf16 A/B WMMA fragment from two 16-byte chunks
// (VGPRs 0..3 = K in [base, base+8), VGPRs 4..7 = K in [base+16, base+24))
__device__ __forceinline__ v16bf ld_frag(const u16* p0, const u16* p1) {
    AB t;
    t.lo = *(const U4*)p0;
    t.hi = *(const U4*)p1;
    return __builtin_bit_cast(v16bf, t);
}

__device__ __forceinline__ v8f wmma_bf16(v16bf a, v16bf b, v8f c) {
    return __builtin_amdgcn_wmma_f32_16x16x32_bf16(
        /*neg_a=*/false, a, /*neg_b=*/false, b,
        /*c_mod=*/(short)0, c, /*reuse_a=*/false, /*reuse_b=*/false);
}

#define EMB  1024
#define HSZ  64
#define TT   2048
#define BB   8

// ---------------------------------------------------------------------------
// Kernel 1: q = x@Wq, k = x@Wk, v = x@Wv.
//   q_ws, k_ws: [B*T][64] bf16 (row major)
//   vT_ws:      [B][64][T]  bf16 (v transposed, for V^T A-fragments)
// 128 rows of x per block, 256 threads = 8 waves; each wave owns a 16-row
// tile. K-chunk = 64: 24 WMMAs per barrier period; LDS = 16KB + 24KB.
// W staged transposed to [n][k] with paired-k u32 stores.
// ---------------------------------------------------------------------------
__global__ __launch_bounds__(256) void proj_kernel(
    const float* __restrict__ x,
    const float* __restrict__ Wq, const float* __restrict__ Wk,
    const float* __restrict__ Wv,
    u16* __restrict__ q_ws, u16* __restrict__ k_ws, u16* __restrict__ vT_ws)
{
    __shared__ u16 xs[128 * 64];          // 16 KB : x chunk  [row][k] bf16
    __shared__ u32 wsh32[3 * 64 * 32];    // 24 KB : W chunks [p][n][k/2] packed
    u16* wsh = (u16*)wsh32;               //          == [p][n][k] bf16

    const int rowBase = blockIdx.x * 128;
    const int tid  = threadIdx.x;
    const int w    = tid >> 5;
    const int lane = tid & 31;
    const int ln16 = lane & 15;
    const int kh   = lane >> 4;

    const float* Wp[3] = { Wq, Wk, Wv };

    v8f acc[3][4];
#pragma unroll
    for (int p = 0; p < 3; ++p)
#pragma unroll
        for (int nt = 0; nt < 4; ++nt)
            acc[p][nt] = (v8f){0.f,0.f,0.f,0.f,0.f,0.f,0.f,0.f};

#pragma unroll 1
    for (int kc = 0; kc < EMB / 64; ++kc) {
        const int kbase = kc * 64;

        // stage x chunk: 128x64 f32 -> bf16 LDS (float4 per thread x8)
#pragma unroll
        for (int i = 0; i < 8; ++i) {
            int idx4 = tid + i * 256;
            int lin  = idx4 * 4;
            int row  = lin >> 6;
            int col  = lin & 63;
            const float4 v = *(const float4*)(x + (size_t)(rowBase + row) * EMB + kbase + col);
            U2 pk;
            pk.a = f2bf2(v.x, v.y);
            pk.b = f2bf2(v.z, v.w);
            *(U2*)&xs[row * 64 + col] = pk;
        }
        // stage W chunks transposed to [n][k], two k's per u32 LDS store
#pragma unroll
        for (int p = 0; p < 3; ++p) {
#pragma unroll
            for (int i = 0; i < 8; ++i) {
                int e  = tid + i * 256;       // 0..2047 pair index
                int n  = e & 63;              // consecutive threads -> coalesced
                int kp = e >> 6;              // 0..31
                const float w0 = Wp[p][(size_t)(kbase + 2 * kp)     * HSZ + n];
                const float w1 = Wp[p][(size_t)(kbase + 2 * kp + 1) * HSZ + n];
                wsh32[p * 2048 + n * 32 + kp] = f2bf2(w0, w1);
            }
        }
        __syncthreads();

        // A fragments: this wave's 16 rows, K halves 0..31 and 32..63
        const u16* ap = &xs[(w * 16 + ln16) * 64 + kh * 8];
        v16bf a0 = ld_frag(ap,      ap + 16);
        v16bf a1 = ld_frag(ap + 32, ap + 48);

#pragma unroll
        for (int p = 0; p < 3; ++p) {
#pragma unroll
            for (int nt = 0; nt < 4; ++nt) {
                const u16* bp = &wsh[p * 4096 + (nt * 16 + ln16) * 64 + kh * 8];
                v16bf b0 = ld_frag(bp,      bp + 16);
                v16bf b1 = ld_frag(bp + 32, bp + 48);
                acc[p][nt] = wmma_bf16(a1, b1, wmma_bf16(a0, b0, acc[p][nt]));
            }
        }
        __syncthreads();
    }

    // Write results. C layout: vgpr j -> M = j + 8*kh; N = ln16.
    const int r0 = rowBase + w * 16;
#pragma unroll
    for (int p = 0; p < 3; ++p) {
#pragma unroll
        for (int nt = 0; nt < 4; ++nt) {
            const int h = nt * 16 + ln16;
#pragma unroll
            for (int j = 0; j < 8; ++j) {
                const int r = r0 + j + 8 * kh;
                const u16 bf = f2bf(acc[p][nt][j]);
                if (p == 0)      q_ws[(size_t)r * HSZ + h] = bf;
                else if (p == 1) k_ws[(size_t)r * HSZ + h] = bf;
                else {
                    const int bidx = r >> 11, t = r & (TT - 1);
                    vT_ws[((size_t)bidx * HSZ + h) * TT + t] = bf;
                }
            }
        }
    }
}

// ---------------------------------------------------------------------------
// Kernel 2: flash attention with swapped roles (scores = k . q^T, softmax over s):
//   out[t,:] = sum_{s<=t} softmax_s(k[t].q[s] * 0.125) * v[s]
// Computes S^T = q_tile @ k^T so the C layout (lane = column t) doubles as the
// P^T B-fragment layout for O^T = V^T @ P^T: no cross-lane transpose needed.
// Block = 128 threads (4 waves), each wave owns 16 t-rows; grid (T/64, B).
// No LDS: q/k/vT (6 MB bf16 total) are L2-resident on MI455X (192 MB L2).
// ---------------------------------------------------------------------------
__global__ __launch_bounds__(128) void attn_kernel(
    const u16* __restrict__ q_ws, const u16* __restrict__ k_ws,
    const u16* __restrict__ vT_ws, float* __restrict__ out)
{
    const int b     = blockIdx.y;
    const int w     = threadIdx.x >> 5;
    const int lane  = threadIdx.x & 31;
    const int ln16  = lane & 15;
    const int kh    = lane >> 4;
    const int tbase = blockIdx.x * 64 + w * 16;
    const int t     = tbase + ln16;            // this lane's column (t index)

    // Preload K B-fragments for this wave's 16 t's (loop invariant): h = 0..63
    const u16* kp = k_ws + ((size_t)b * TT + t) * HSZ;
    const v16bf kf0 = ld_frag(kp +  0 + kh * 8, kp + 16 + kh * 8);
    const v16bf kf1 = ld_frag(kp + 32 + kh * 8, kp + 48 + kh * 8);

    v8f o[4];
#pragma unroll
    for (int ht = 0; ht < 4; ++ht)
        o[ht] = (v8f){0.f,0.f,0.f,0.f,0.f,0.f,0.f,0.f};
    float m = -3.0e38f, lsum = 0.f;

    const int nchunks = (tbase + 16 + 31) >> 5;   // s chunks of 32, up to diagonal
#pragma unroll 1
    for (int c = 0; c < nchunks; ++c) {
        const int sbase = c * 32;
        const u16* qb = q_ws + ((size_t)b * TT + sbase) * HSZ;
        if (c + 1 < nchunks) __builtin_prefetch(qb + 32 * HSZ, 0, 0);

        // Q A-fragments: rows s = sbase+ln16 and sbase+16+ln16, K = h (64)
        const u16* qr0 = qb + (size_t)ln16 * HSZ;
        const u16* qr1 = qb + (size_t)(16 + ln16) * HSZ;
        v16bf qa00 = ld_frag(qr0 +  0 + kh * 8, qr0 + 16 + kh * 8);
        v16bf qa01 = ld_frag(qr0 + 32 + kh * 8, qr0 + 48 + kh * 8);
        v16bf qa10 = ld_frag(qr1 +  0 + kh * 8, qr1 + 16 + kh * 8);
        v16bf qa11 = ld_frag(qr1 + 32 + kh * 8, qr1 + 48 + kh * 8);

        v8f z = (v8f){0.f,0.f,0.f,0.f,0.f,0.f,0.f,0.f};
        v8f s0 = wmma_bf16(qa01, kf1, wmma_bf16(qa00, kf0, z)); // s rows sbase..+15
        v8f s1 = wmma_bf16(qa11, kf1, wmma_bf16(qa10, kf0, z)); // s rows sbase+16..+31

        // scale + causal mask (per-lane data select; EXEC untouched)
        float p0[8], p1[8];
        float cmax = -3.0e38f;
#pragma unroll
        for (int j = 0; j < 8; ++j) {
            const int si0 = sbase + j + 8 * kh;
            const int si1 = si0 + 16;
            float a0 = (si0 <= t) ? s0[j] * 0.125f : -3.0e38f;
            float a1 = (si1 <= t) ? s1[j] * 0.125f : -3.0e38f;
            p0[j] = a0; p1[j] = a1;
            cmax = fmaxf(cmax, fmaxf(a0, a1));
        }
        cmax = fmaxf(cmax, __shfl_xor(cmax, 16, 32));   // combine lane halves
        const float mnew = fmaxf(m, cmax);
        const float fac  = __expf(m - mnew);
        m = mnew;
        lsum *= fac;
#pragma unroll
        for (int ht = 0; ht < 4; ++ht) o[ht] *= fac;

        // P^T B-fragment: lane's own values land exactly in the right slots
        H16 ph;
        float rsum = 0.f;
#pragma unroll
        for (int j = 0; j < 8; ++j) {
            const float e0 = __expf(p0[j] - m);
            const float e1 = __expf(p1[j] - m);
            rsum += e0 + e1;
            ph.h[j]     = f2bf(e0);
            ph.h[8 + j] = f2bf(e1);
        }
        rsum += __shfl_xor(rsum, 16, 32);
        lsum += rsum;
        const v16bf pb = __builtin_bit_cast(v16bf, ph);

        // O^T += V^T @ P^T  (A = V^T rows h, K = s chunk of 32)
#pragma unroll
        for (int ht = 0; ht < 4; ++ht) {
            const u16* vr = vT_ws + ((size_t)b * HSZ + ht * 16 + ln16) * TT + sbase;
            v16bf va = ld_frag(vr + kh * 8, vr + 16 + kh * 8);
            o[ht] = wmma_bf16(va, pb, o[ht]);
        }
    }

    // Final normalize + store: O^T C layout -> out[b][t][h] f32
    const float inv = 1.0f / lsum;
#pragma unroll
    for (int ht = 0; ht < 4; ++ht) {
#pragma unroll
        for (int j = 0; j < 8; ++j) {
            const int h = ht * 16 + j + 8 * kh;
            out[((size_t)b * TT + t) * HSZ + h] = o[ht][j] * inv;
        }
    }
}

extern "C" void kernel_launch(void* const* d_in, const int* in_sizes, int n_in,
                              void* d_out, int out_size, void* d_ws, size_t ws_size,
                              hipStream_t stream) {
    const float* x  = (const float*)d_in[0];
    const float* Wq = (const float*)d_in[1];
    const float* Wk = (const float*)d_in[2];
    const float* Wv = (const float*)d_in[3];

    // workspace: q (2MB) + k (2MB) + vT (2MB) bf16
    u16* q_ws  = (u16*)d_ws;
    u16* k_ws  = q_ws + (size_t)BB * TT * HSZ;
    u16* vT_ws = k_ws + (size_t)BB * TT * HSZ;

    proj_kernel<<<(BB * TT) / 128, 256, 0, stream>>>(x, Wq, Wk, Wv, q_ws, k_ws, vT_ws);
    attn_kernel<<<dim3(TT / 64, BB), 128, 0, stream>>>(q_ws, k_ws, vT_ws, (float*)d_out);
}
